// MultiHeadAttention_87582973100767
// MI455X (gfx1250) — matrix operand order repaired and checked by reading, not compiled, not run on hardware
//
#include <hip/hip_runtime.h>

// MI455X / gfx1250 fused multi-head attention forward.
// fp32 in/out, bf16 WMMA (v_wmma_f32_16x16x32_bf16) internally, fp32 accum.
// Double-buffered LDS GEMM pipeline + global_prefetch in the flash loop.
//
// B=2, S=2048, D=1024, H=16, DK=64.

#define BB   2
#define SS   2048
#define DDIM 1024
#define HH   16
#define DKK  64
#define BSROWS (BB * SS)   // 4096

typedef __attribute__((ext_vector_type(16))) __bf16       v16bf;
typedef __attribute__((ext_vector_type(8)))  float        v8f;
typedef __attribute__((ext_vector_type(4)))  float        v4f;
typedef __attribute__((ext_vector_type(4)))  unsigned int v4u;

union FragBf {               // 16 bf16 = 32 bytes = one WMMA A/B operand
  v16bf h;
  v4u   u4[2];
};

// fp32 -> bf16, round to nearest even
__device__ __forceinline__ unsigned short f2bf(float f) {
  unsigned int u = __builtin_bit_cast(unsigned int, f);
  u += 0x7fffu + ((u >> 16) & 1u);
  return (unsigned short)(u >> 16);
}
__device__ __forceinline__ unsigned int pk2(float lo, float hi) {
  return (unsigned int)f2bf(lo) | ((unsigned int)f2bf(hi) << 16);
}

// ---------------------------------------------------------------------------
// Kernel 1: projection GEMM  dst[m][n] = sum_k X[m][k] * W[n][k] + bias[n]
//   X: [4096,1024] fp32 (q/k/v flattened over B*S), W: [1024,1024] fp32
//   dst: bf16, layout [B,H,S,DK] (vtran=0) or [B,H,DK,S] (vtran=1, for V)
// Block: 256 thr (8 waves), tile 128x128, wave tile 64x32, K-step 32.
// Double-buffered LDS: global loads of tile t+1 overlap the 8 WMMAs of tile t.
// ---------------------------------------------------------------------------
__global__ __launch_bounds__(256) void proj_gemm_bf16(
    const float* __restrict__ X, const float* __restrict__ W,
    const float* __restrict__ bias, unsigned short* __restrict__ dst,
    int vtran)
{
  __shared__ __align__(16) unsigned short Ab[2][128 * 32];   // 2 x 8KB
  __shared__ __align__(16) unsigned short Bb[2][128 * 32];   // 2 x 8KB

  const int tid  = threadIdx.x;
  const int m0   = blockIdx.y * 128;
  const int n0   = blockIdx.x * 128;
  const int lane = tid & 31, wid = tid >> 5;
  const int ln   = lane & 15, g = lane >> 4;          // half-wave group
  const int wm0  = (wid & 1) * 64;                    // 2 wave rows
  const int wn0  = (wid >> 1) * 32;                   // 4 wave cols

  v8f z;
#pragma unroll
  for (int i = 0; i < 8; ++i) z[i] = 0.0f;

  v8f acc[4][2];
#pragma unroll
  for (int mi = 0; mi < 4; ++mi)
#pragma unroll
    for (int ni = 0; ni < 2; ++ni) acc[mi][ni] = z;

  const int srow = tid >> 1, shalf = tid & 1;         // staging: 128 rows x 2 halves
  const float* xa = X + (size_t)(m0 + srow) * DDIM + shalf * 16;
  const float* xb = W + (size_t)(n0 + srow) * DDIM + shalf * 16;

  // prologue: load tile 0 into registers
  v4f ra[4], rb[4];
#pragma unroll
  for (int i = 0; i < 4; ++i) { ra[i] = ((const v4f*)xa)[i]; rb[i] = ((const v4f*)xb)[i]; }

  const int NT = DDIM / 32;   // 32 K-tiles
  for (int t = 0; t < NT; ++t) {
    const int buf = t & 1;
    // convert + store tile t into LDS[buf]
    {
      v4u c0 = { pk2(ra[0][0],ra[0][1]), pk2(ra[0][2],ra[0][3]), pk2(ra[1][0],ra[1][1]), pk2(ra[1][2],ra[1][3]) };
      v4u c1 = { pk2(ra[2][0],ra[2][1]), pk2(ra[2][2],ra[2][3]), pk2(ra[3][0],ra[3][1]), pk2(ra[3][2],ra[3][3]) };
      *(v4u*)&Ab[buf][srow * 32 + shalf * 16]     = c0;
      *(v4u*)&Ab[buf][srow * 32 + shalf * 16 + 8] = c1;
      v4u d0 = { pk2(rb[0][0],rb[0][1]), pk2(rb[0][2],rb[0][3]), pk2(rb[1][0],rb[1][1]), pk2(rb[1][2],rb[1][3]) };
      v4u d1 = { pk2(rb[2][0],rb[2][1]), pk2(rb[2][2],rb[2][3]), pk2(rb[3][0],rb[3][1]), pk2(rb[3][2],rb[3][3]) };
      *(v4u*)&Bb[buf][srow * 32 + shalf * 16]     = d0;
      *(v4u*)&Bb[buf][srow * 32 + shalf * 16 + 8] = d1;
    }
    __syncthreads();
    // issue global loads for tile t+1 (in flight during the WMMAs below)
    if (t + 1 < NT) {
      const v4f* pa = (const v4f*)(xa + (t + 1) * 32);
      const v4f* pb = (const v4f*)(xb + (t + 1) * 32);
#pragma unroll
      for (int i = 0; i < 4; ++i) { ra[i] = pa[i]; rb[i] = pb[i]; }
    }
    // compute tile t from LDS[buf]
    FragBf fa[4], fb[2];
#pragma unroll
    for (int mi = 0; mi < 4; ++mi) {
      const int r = (wm0 + mi * 16 + ln) * 32;
      fa[mi].u4[0] = *(const v4u*)&Ab[buf][r + g * 8];        // K = g*8..g*8+7
      fa[mi].u4[1] = *(const v4u*)&Ab[buf][r + 16 + g * 8];   // K = 16+g*8..
    }
#pragma unroll
    for (int ni = 0; ni < 2; ++ni) {
      const int r = (wn0 + ni * 16 + ln) * 32;                // K = g*8..g*8+7, 16+g*8..
      fb[ni].u4[0] = *(const v4u*)&Bb[buf][r + g * 8];
      fb[ni].u4[1] = *(const v4u*)&Bb[buf][r + 16 + g * 8];
    }
#pragma unroll
    for (int mi = 0; mi < 4; ++mi)
#pragma unroll
      for (int ni = 0; ni < 2; ++ni)
        acc[mi][ni] = __builtin_amdgcn_wmma_f32_16x16x32_bf16(
            false, fa[mi].h, false, fb[ni].h, (short)0, acc[mi][ni], false, false);
  }

  // epilogue: bias + bf16 store into head layout
#pragma unroll
  for (int mi = 0; mi < 4; ++mi) {
#pragma unroll
    for (int ni = 0; ni < 2; ++ni) {
      const int gn = n0 + wn0 + ni * 16 + ln;
      const float bv = bias[gn];
      const int h = gn >> 6, dk = gn & 63;
#pragma unroll
      for (int r = 0; r < 8; ++r) {
        const int gm = m0 + wm0 + mi * 16 + r + 8 * g;   // C layout: M = r + 8*g
        const int bI = gm >> 11, s = gm & 2047;
        const float val = acc[mi][ni][r] + bv;
        const size_t idx = vtran
            ? ((size_t)(bI * HH + h) * DKK + dk) * SS + s    // V transposed [B,H,DK,S]
            : ((size_t)(bI * HH + h) * SS + s) * DKK + dk;   // [B,H,S,DK]
        dst[idx] = f2bf(val);
      }
    }
  }
}

// ---------------------------------------------------------------------------
// Kernel 2: causal flash attention, one (b,h) per blockIdx.y, 128 q-rows per
// block, 16 q-rows per wave. 32-key steps: QK^T = 2 N-frags x K=64 chain
// (4 WMMA), online softmax via 16-lane shfl_xor row reductions, P bounced
// through per-wave LDS (C-layout -> A-layout), PV = 4 WMMA against V^T.
// Next j-block's K/V lines are prefetched (global_prefetch_b8) one step ahead.
// ---------------------------------------------------------------------------
__global__ __launch_bounds__(256) void flash_attn_bf16(
    const unsigned short* __restrict__ Qh, const unsigned short* __restrict__ Kh,
    const unsigned short* __restrict__ Vt, unsigned short* __restrict__ ctx)
{
  __shared__ __align__(16) unsigned short Plds[8 * 16 * 32];  // 8KB, per-wave slices

  const int tid  = threadIdx.x;
  const int lane = tid & 31, wid = tid >> 5;
  const int ln   = lane & 15, g = lane >> 4;
  const int bh   = blockIdx.y;
  const int q0   = blockIdx.x * 128 + wid * 16;

  const unsigned short* Qb = Qh + (size_t)bh * SS * DKK;
  const unsigned short* Kb = Kh + (size_t)bh * SS * DKK;
  const unsigned short* Vb = Vt + (size_t)bh * DKK * SS;
  unsigned short* Pw = &Plds[wid * (16 * 32)];

  v8f z;
#pragma unroll
  for (int i = 0; i < 8; ++i) z[i] = 0.0f;

  // Q fragments (16 rows x 64 d), resident for the whole kernel
  FragBf qa[2];
#pragma unroll
  for (int kf = 0; kf < 2; ++kf) {
    const size_t base = (size_t)(q0 + ln) * DKK + kf * 32;
    qa[kf].u4[0] = *(const v4u*)(Qb + base + g * 8);
    qa[kf].u4[1] = *(const v4u*)(Qb + base + 16 + g * 8);
  }

  v8f o[4];
#pragma unroll
  for (int nf = 0; nf < 4; ++nf) o[nf] = z;
  float m[8], l[8];
#pragma unroll
  for (int r = 0; r < 8; ++r) { m[r] = -__builtin_inff(); l[r] = 0.0f; }

  const int nsteps = (q0 + 16 + 31) >> 5;   // causal: keys 0 .. q0+15
  for (int js = 0; js < nsteps; ++js) {
    const int j0 = js * 32;

    // ---- prefetch next j-block (K: one key row per lane; V^T: 2 rows/lane) ----
    if (js + 1 < nsteps) {
      const int jn0 = j0 + 32;
      __builtin_prefetch(Kb + (size_t)(jn0 + lane) * DKK, 0, 0);
      __builtin_prefetch(Vb + (size_t)lane * SS + jn0, 0, 0);
      __builtin_prefetch(Vb + (size_t)(lane + 32) * SS + jn0, 0, 0);
    }

    // ---- scores S = Q K^T / sqrt(DK), two 16x16 frags ----
    v8f sc[2];
#pragma unroll
    for (int jn = 0; jn < 2; ++jn) {
      FragBf kb0, kb1;
      const size_t kbase = (size_t)(j0 + jn * 16 + ln) * DKK;   // column = key row of Kh
      kb0.u4[0] = *(const v4u*)(Kb + kbase + g * 8);
      kb0.u4[1] = *(const v4u*)(Kb + kbase + 16 + g * 8);
      kb1.u4[0] = *(const v4u*)(Kb + kbase + 32 + g * 8);
      kb1.u4[1] = *(const v4u*)(Kb + kbase + 32 + 16 + g * 8);
      v8f s = __builtin_amdgcn_wmma_f32_16x16x32_bf16(false, qa[0].h, false, kb0.h,
                                                      (short)0, z, false, false);
      s = __builtin_amdgcn_wmma_f32_16x16x32_bf16(false, qa[1].h, false, kb1.h,
                                                  (short)0, s, false, false);
      sc[jn] = s;
    }

    // ---- scale + causal mask (C layout: row q0+r+8g, col j0+jn*16+ln) ----
#pragma unroll
    for (int jn = 0; jn < 2; ++jn)
#pragma unroll
      for (int r = 0; r < 8; ++r) {
        const int qrow = q0 + r + 8 * g;
        const int kcol = j0 + jn * 16 + ln;
        const float s = sc[jn][r] * 0.125f;   // 1/sqrt(64)
        sc[jn][r] = (kcol > qrow) ? -__builtin_inff() : s;
      }

    // ---- online softmax: row max / rescale / row sum ----
    float mnew[8], alpha[8];
#pragma unroll
    for (int r = 0; r < 8; ++r) {
      float cur = fmaxf(sc[0][r], sc[1][r]);
      cur = fmaxf(cur, __shfl_xor(cur, 1, 32));
      cur = fmaxf(cur, __shfl_xor(cur, 2, 32));
      cur = fmaxf(cur, __shfl_xor(cur, 4, 32));
      cur = fmaxf(cur, __shfl_xor(cur, 8, 32));   // stays within 16-lane half
      mnew[r]  = fmaxf(m[r], cur);
      alpha[r] = __expf(m[r] - mnew[r]);
      m[r]     = mnew[r];
    }
#pragma unroll
    for (int jn = 0; jn < 2; ++jn)
#pragma unroll
      for (int r = 0; r < 8; ++r)
        sc[jn][r] = __expf(sc[jn][r] - mnew[r]);
#pragma unroll
    for (int r = 0; r < 8; ++r) {
      float rs = sc[0][r] + sc[1][r];
      rs += __shfl_xor(rs, 1, 32);
      rs += __shfl_xor(rs, 2, 32);
      rs += __shfl_xor(rs, 4, 32);
      rs += __shfl_xor(rs, 8, 32);
      l[r] = l[r] * alpha[r] + rs;
    }

    // ---- P: C-layout -> LDS row-major 16x32 -> reload as A-layout frag ----
#pragma unroll
    for (int jn = 0; jn < 2; ++jn)
#pragma unroll
      for (int r = 0; r < 8; ++r)
        Pw[(r + 8 * g) * 32 + jn * 16 + ln] = f2bf(sc[jn][r]);
    asm volatile("s_wait_dscnt 0" ::: "memory");   // per-wave LDS RAW fence
    FragBf pf;
    pf.u4[0] = *(const v4u*)&Pw[ln * 32 + g * 8];
    pf.u4[1] = *(const v4u*)&Pw[ln * 32 + 16 + g * 8];

    // ---- O = diag(alpha) O + P V  (V^T layout gives K-contiguous B frags) ----
#pragma unroll
    for (int nf = 0; nf < 4; ++nf) {
      FragBf vf;
      const size_t vbase = (size_t)(nf * 16 + ln) * SS + j0;
      vf.u4[0] = *(const v4u*)(Vb + vbase + g * 8);
      vf.u4[1] = *(const v4u*)(Vb + vbase + 16 + g * 8);
      v8f on = o[nf];
#pragma unroll
      for (int r = 0; r < 8; ++r) on[r] *= alpha[r];
      o[nf] = __builtin_amdgcn_wmma_f32_16x16x32_bf16(false, pf.h, false, vf.h,
                                                      (short)0, on, false, false);
    }
  }

  // ---- normalize and store context bf16 [B,S,D] ----
  const int b = bh >> 4, h = bh & 15;
#pragma unroll
  for (int nf = 0; nf < 4; ++nf) {
#pragma unroll
    for (int r = 0; r < 8; ++r) {
      const int q = q0 + r + 8 * g;
      const int dcol = h * DKK + nf * 16 + ln;
      const float val = o[nf][r] / l[r];
      ctx[(size_t)(b * SS + q) * DDIM + dcol] = f2bf(val);
    }
  }
}

// ---------------------------------------------------------------------------
// Kernel 3: output projection  out[m][n] = sum_k ctx[m][k] * Wo[n][k] + bo[n]
// ctx already bf16; Wo converted on stage-in; fp32 output. Same double-buffered
// pipeline as kernel 1.
// ---------------------------------------------------------------------------
__global__ __launch_bounds__(256) void out_proj_f32(
    const unsigned short* __restrict__ Ctx, const float* __restrict__ W,
    const float* __restrict__ bias, float* __restrict__ out)
{
  __shared__ __align__(16) unsigned short Ab[2][128 * 32];
  __shared__ __align__(16) unsigned short Bb[2][128 * 32];

  const int tid  = threadIdx.x;
  const int m0   = blockIdx.y * 128;
  const int n0   = blockIdx.x * 128;
  const int lane = tid & 31, wid = tid >> 5;
  const int ln   = lane & 15, g = lane >> 4;
  const int wm0  = (wid & 1) * 64;
  const int wn0  = (wid >> 1) * 32;

  v8f z;
#pragma unroll
  for (int i = 0; i < 8; ++i) z[i] = 0.0f;
  v8f acc[4][2];
#pragma unroll
  for (int mi = 0; mi < 4; ++mi)
#pragma unroll
    for (int ni = 0; ni < 2; ++ni) acc[mi][ni] = z;

  const int srow = tid >> 1, shalf = tid & 1;
  const unsigned short* xa = Ctx + (size_t)(m0 + srow) * DDIM + shalf * 16;
  const float*          xb = W   + (size_t)(n0 + srow) * DDIM + shalf * 16;

  v4u ra[2];
  v4f rb[4];
#pragma unroll
  for (int i = 0; i < 2; ++i) ra[i] = ((const v4u*)xa)[i];
#pragma unroll
  for (int i = 0; i < 4; ++i) rb[i] = ((const v4f*)xb)[i];

  const int NT = DDIM / 32;
  for (int t = 0; t < NT; ++t) {
    const int buf = t & 1;
    {
      *(v4u*)&Ab[buf][srow * 32 + shalf * 16]     = ra[0];
      *(v4u*)&Ab[buf][srow * 32 + shalf * 16 + 8] = ra[1];
      v4u d0 = { pk2(rb[0][0],rb[0][1]), pk2(rb[0][2],rb[0][3]), pk2(rb[1][0],rb[1][1]), pk2(rb[1][2],rb[1][3]) };
      v4u d1 = { pk2(rb[2][0],rb[2][1]), pk2(rb[2][2],rb[2][3]), pk2(rb[3][0],rb[3][1]), pk2(rb[3][2],rb[3][3]) };
      *(v4u*)&Bb[buf][srow * 32 + shalf * 16]     = d0;
      *(v4u*)&Bb[buf][srow * 32 + shalf * 16 + 8] = d1;
    }
    __syncthreads();
    if (t + 1 < NT) {
      const v4u* pa = (const v4u*)(xa + (t + 1) * 32);
      const v4f* pb = (const v4f*)(xb + (t + 1) * 32);
      ra[0] = pa[0]; ra[1] = pa[1];
#pragma unroll
      for (int i = 0; i < 4; ++i) rb[i] = pb[i];
    }
    FragBf fa[4], fb[2];
#pragma unroll
    for (int mi = 0; mi < 4; ++mi) {
      const int r = (wm0 + mi * 16 + ln) * 32;
      fa[mi].u4[0] = *(const v4u*)&Ab[buf][r + g * 8];
      fa[mi].u4[1] = *(const v4u*)&Ab[buf][r + 16 + g * 8];
    }
#pragma unroll
    for (int ni = 0; ni < 2; ++ni) {
      const int r = (wn0 + ni * 16 + ln) * 32;
      fb[ni].u4[0] = *(const v4u*)&Bb[buf][r + g * 8];
      fb[ni].u4[1] = *(const v4u*)&Bb[buf][r + 16 + g * 8];
    }
#pragma unroll
    for (int mi = 0; mi < 4; ++mi)
#pragma unroll
      for (int ni = 0; ni < 2; ++ni)
        acc[mi][ni] = __builtin_amdgcn_wmma_f32_16x16x32_bf16(
            false, fa[mi].h, false, fb[ni].h, (short)0, acc[mi][ni], false, false);
  }

#pragma unroll
  for (int mi = 0; mi < 4; ++mi) {
#pragma unroll
    for (int ni = 0; ni < 2; ++ni) {
      const int gn = n0 + wn0 + ni * 16 + ln;
      const float bv = bias[gn];
#pragma unroll
      for (int r = 0; r < 8; ++r) {
        const int gm = m0 + wm0 + mi * 16 + r + 8 * g;
        out[(size_t)gm * DDIM + gn] = acc[mi][ni][r] + bv;
      }
    }
  }
}

// ---------------------------------------------------------------------------
extern "C" void kernel_launch(void* const* d_in, const int* in_sizes, int n_in,
                              void* d_out, int out_size, void* d_ws, size_t ws_size,
                              hipStream_t stream) {
  const float* q   = (const float*)d_in[0];
  const float* k   = (const float*)d_in[1];
  const float* v   = (const float*)d_in[2];
  // d_in[3] = causal mask, statically known -> unused
  const float* w_q = (const float*)d_in[4];
  const float* b_q = (const float*)d_in[5];
  const float* w_k = (const float*)d_in[6];
  const float* b_k = (const float*)d_in[7];
  const float* w_v = (const float*)d_in[8];
  const float* b_v = (const float*)d_in[9];
  const float* w_o = (const float*)d_in[10];
  const float* b_o = (const float*)d_in[11];
  float* out = (float*)d_out;

  // workspace: 4 x 4M bf16 = 32 MB, all fully written before read
  const size_t nh = (size_t)BB * HH * SS * DKK;       // 4,194,304 elements
  unsigned short* Qh  = (unsigned short*)d_ws;        // [B,H,S,DK]
  unsigned short* Kh  = Qh + nh;                      // [B,H,S,DK]
  unsigned short* Vt  = Kh + nh;                      // [B,H,DK,S]
  unsigned short* Ctx = Vt + nh;                      // [B,S,D]
  (void)in_sizes; (void)n_in; (void)out_size; (void)ws_size;

  dim3 blk(256);
  dim3 gproj(DDIM / 128, BSROWS / 128);               // (8, 32)
  proj_gemm_bf16<<<gproj, blk, 0, stream>>>(q, w_q, b_q, Qh, 0);
  proj_gemm_bf16<<<gproj, blk, 0, stream>>>(k, w_k, b_k, Kh, 0);
  proj_gemm_bf16<<<gproj, blk, 0, stream>>>(v, w_v, b_v, Vt, 1);

  dim3 gattn(SS / 128, BB * HH);                      // (16, 32)
  flash_attn_bf16<<<gattn, blk, 0, stream>>>(Qh, Kh, Vt, Ctx);

  out_proj_f32<<<gproj, blk, 0, stream>>>(Ctx, w_o, b_o, out);
}
